// SMoE_56324201120513
// MI455X (gfx1250) — matrix operand order repaired and checked by reading, not compiled, not run on hardware
//
#include <hip/hip_runtime.h>

// ---------------------------------------------------------------------------
// Types for CDNA5 WMMA (wave32, v_wmma_f32_16x16x32_bf16)
// ---------------------------------------------------------------------------
typedef __attribute__((ext_vector_type(16))) __bf16 v16bf;
typedef __attribute__((ext_vector_type(8)))  __bf16 v8bf;
typedef __attribute__((ext_vector_type(8)))  float  v8f;

static __device__ __forceinline__ unsigned short f2bf_bits(float f) {
    unsigned u = __float_as_uint(f);
    unsigned r = (u + 0x7FFFu + ((u >> 16) & 1u)) >> 16;  // round-to-nearest-even
    return (unsigned short)r;
}

// Fast SiLU: x * rcp(1 + exp(-x)) using v_rcp_f32 instead of IEEE division.
static __device__ __forceinline__ float fast_silu(float x) {
    return x * __builtin_amdgcn_rcpf(1.f + __expf(-x));
}

#define CONCAT16(LO, HI) __builtin_shufflevector((LO), (HI), 0,1,2,3,4,5,6,7,8,9,10,11,12,13,14,15)

// ---------------------------------------------------------------------------
// CDNA5 async global->LDS copy (ASYNCcnt-tracked, bypasses VGPRs).
// Builtin signature (from compiler diagnostics):
//   void __builtin_amdgcn_global_load_async_to_lds_b128(
//       int4 addrspace(1)* src, int4 addrspace(3)* dst, imm int off, imm int cpol)
// Gated on builtin availability; falls back to sync copies otherwise.
// ---------------------------------------------------------------------------
#if defined(__HIP_DEVICE_COMPILE__) && defined(__gfx1250__) && \
    __has_builtin(__builtin_amdgcn_global_load_async_to_lds_b128)
#define MOE_ASYNC 1
#else
#define MOE_ASYNC 0
#endif

#if MOE_ASYNC
typedef int moe_v4i __attribute__((vector_size(16)));
#define AS1CAST(p) ((__attribute__((address_space(1))) moe_v4i*)(p))
#define AS3CAST(p) ((__attribute__((address_space(3))) moe_v4i*)(p))
static __device__ __forceinline__ void moe_wait_async0() {
#if __has_builtin(__builtin_amdgcn_s_wait_asynccnt)
    __builtin_amdgcn_s_wait_asynccnt(0);
#else
    asm volatile("s_wait_asynccnt 0x0" ::: "memory");
#endif
}
#else
static __device__ __forceinline__ void moe_wait_async0() {}
#endif

// ---------------------------------------------------------------------------
// 1) Transpose + f32 -> bf16 convert:  in  (R x C) f32  ->  out (C x R) bf16
//    grid: (C/32, R/32, E); per-expert contiguous R*C blocks.
// ---------------------------------------------------------------------------
__global__ __launch_bounds__(256) void smoe_cvt_t(const float* __restrict__ in,
                                                  unsigned short* __restrict__ out,
                                                  int R, int C) {
    __shared__ float tile[32][33];
    const int e = blockIdx.z;
    in  += (size_t)e * R * C;
    out += (size_t)e * R * C;
    const int cb = blockIdx.x * 32, rb = blockIdx.y * 32;
    const int tx = threadIdx.x & 31, ty = threadIdx.x >> 5;   // ty in 0..7
#pragma unroll
    for (int i = 0; i < 4; ++i)
        tile[ty + i * 8][tx] = in[(size_t)(rb + ty + i * 8) * C + cb + tx];
    __syncthreads();
#pragma unroll
    for (int i = 0; i < 4; ++i)
        out[(size_t)(cb + ty + i * 8) * R + rb + tx] = f2bf_bits(tile[tx][ty + i * 8]);
}

// ---------------------------------------------------------------------------
// 2) Routing: one wave per token. logits = x@Wg + bg ; argmax ; softmax gate.
// ---------------------------------------------------------------------------
__global__ __launch_bounds__(256) void smoe_route(const float* __restrict__ x,
                                                  const float* __restrict__ Wg,
                                                  const float* __restrict__ bg,
                                                  int* __restrict__ eid,
                                                  float* __restrict__ gate,
                                                  int N, int D) {
    const int lane  = threadIdx.x & 31;
    const int token = blockIdx.x * 8 + (threadIdx.x >> 5);
    if (token >= N) return;
    const float* xr = x + (size_t)token * D;
    float acc[8];
#pragma unroll
    for (int e = 0; e < 8; ++e) acc[e] = 0.f;
    for (int j = lane; j < D; j += 32) {
        float xv = xr[j];
        const float4* w = (const float4*)(Wg + (size_t)j * 8);
        float4 w0 = w[0], w1 = w[1];
        acc[0] += xv * w0.x; acc[1] += xv * w0.y; acc[2] += xv * w0.z; acc[3] += xv * w0.w;
        acc[4] += xv * w1.x; acc[5] += xv * w1.y; acc[6] += xv * w1.z; acc[7] += xv * w1.w;
    }
#pragma unroll
    for (int e = 0; e < 8; ++e) {
#pragma unroll
        for (int off = 16; off > 0; off >>= 1) acc[e] += __shfl_xor(acc[e], off, 32);
        acc[e] += bg[e];
    }
    float m = acc[0]; int best = 0;
#pragma unroll
    for (int e = 1; e < 8; ++e) if (acc[e] > m) { m = acc[e]; best = e; }
    float s = 0.f;
#pragma unroll
    for (int e = 0; e < 8; ++e) s += __expf(acc[e] - m);
    if (lane == 0) { eid[token] = best; gate[token] = 1.f / s; }
}

// ---------------------------------------------------------------------------
// 3) Capacity scan: single block of 1024 threads, deterministic token-order
//    rank per expert via wave ballots + cross-wave prefix. slot=-1 if dropped.
// ---------------------------------------------------------------------------
__global__ __launch_bounds__(1024) void smoe_scan(const int* __restrict__ eid,
                                                  int* __restrict__ slot,
                                                  int N, int cap) {
    __shared__ int wcnt[32][8];
    __shared__ int tot[8];
    __shared__ int running[8];
    const int tid = threadIdx.x, lane = tid & 31, wv = tid >> 5;
    if (tid < 8) running[tid] = 0;
    __syncthreads();
    for (int base = 0; base < N; base += 1024) {
        const int t = base + tid;
        const int e = eid[t];
        const unsigned lt = (1u << lane) - 1u;
        int rank = 0;
#pragma unroll
        for (int ee = 0; ee < 8; ++ee) {
            unsigned b = __builtin_amdgcn_ballot_w32(e == ee);
            if (ee == e) rank = __popc(b & lt);
            if (lane == 0) wcnt[wv][ee] = __popc(b);
        }
        __syncthreads();
        if (tid < 8) {            // exclusive prefix across the 32 waves
            int a = 0;
            for (int w = 0; w < 32; ++w) { int c = wcnt[w][tid]; wcnt[w][tid] = a; a += c; }
            tot[tid] = a;
        }
        __syncthreads();
        const int pos = running[e] + wcnt[wv][e] + rank;
        slot[t] = (pos < cap) ? pos : -1;
        __syncthreads();
        if (tid < 8) running[tid] += tot[tid];
        __syncthreads();
    }
}

// ---------------------------------------------------------------------------
// 4) Scatter tokens to bf16 expert buffers. One wave per token.
// ---------------------------------------------------------------------------
__global__ __launch_bounds__(256) void smoe_scatter(const float* __restrict__ x,
                                                    const int* __restrict__ eid,
                                                    const int* __restrict__ slot,
                                                    unsigned short* __restrict__ buf,
                                                    int N, int D, int cap) {
    const int lane  = threadIdx.x & 31;
    const int token = blockIdx.x * 8 + (threadIdx.x >> 5);
    if (token >= N) return;
    const int s = slot[token];
    if (s < 0) return;
    const int e = eid[token];
    const float* src = x + (size_t)token * D;
    unsigned short* dst = buf + ((size_t)e * cap + s) * D;
    for (int j = lane * 4; j < D; j += 128) {
        float4 v = *(const float4*)(src + j);
        uint2 o;
        o.x = (unsigned)f2bf_bits(v.x) | ((unsigned)f2bf_bits(v.y) << 16);
        o.y = (unsigned)f2bf_bits(v.z) | ((unsigned)f2bf_bits(v.w) << 16);
        *(uint2*)(dst + j) = o;
    }
}

// ---------------------------------------------------------------------------
// WMMA fragment loaders from padded LDS (row stride 40 bf16 = 80 B).
// A layout (16x32 bf16): lane<16 -> K {0..7,16..23}; lane>=16 -> K {8..15,24..31}.
// B layout (32x16 bf16): lane&15 = column; lane<16 -> K {0..7,16..23}; lane>=16 -> K {8..15,24..31}.
// ---------------------------------------------------------------------------
#define LDSPITCH 40

static __device__ __forceinline__ v16bf frag_a(const unsigned short* smem, int row, int hi16) {
    const v8bf* p0 = (const v8bf*)(smem + row * LDSPITCH + hi16 * 8);
    const v8bf* p1 = (const v8bf*)(smem + row * LDSPITCH + hi16 * 8 + 16);
    return CONCAT16(p0[0], p1[0]);
}
static __device__ __forceinline__ v16bf frag_b(const unsigned short* smem, int row, int hi16) {
    const v8bf* p0 = (const v8bf*)(smem + row * LDSPITCH + hi16 * 8);
    const v8bf* p1 = (const v8bf*)(smem + row * LDSPITCH + hi16 * 8 + 16);
    return CONCAT16(p0[0], p1[0]);
}

// ---------------------------------------------------------------------------
// 5) GEMM1: h = silu(binp @ W2) * (binp @ W3).  A: cap x D bf16 (row major),
//    B (pre-transposed): H x D bf16 per expert.  Block tile 128x64, 8 waves,
//    double-buffered LDS with async global->LDS prefetch.
// ---------------------------------------------------------------------------
__global__ __launch_bounds__(256) void smoe_ffn1(const unsigned short* __restrict__ buf,
                                                 const unsigned short* __restrict__ w2t,
                                                 const unsigned short* __restrict__ w3t,
                                                 unsigned short* __restrict__ hbuf,
                                                 int D, int H, int cap) {
    __shared__ __align__(16) unsigned short sA [2][128 * LDSPITCH];
    __shared__ __align__(16) unsigned short sB2[2][64 * LDSPITCH];
    __shared__ __align__(16) unsigned short sB3[2][64 * LDSPITCH];
    const int e = blockIdx.z;
    const int mBase = blockIdx.x * 128;
    const int nBase = blockIdx.y * 64;
    const unsigned short* A  = buf + ((size_t)e * cap + mBase) * D;
    const unsigned short* B2 = w2t + ((size_t)e * H + nBase) * D;
    const unsigned short* B3 = w3t + ((size_t)e * H + nBase) * D;
    const int tid = threadIdx.x, lane = tid & 31, wv = tid >> 5;
    const int mw = (wv & 3) * 32, nw = (wv >> 2) * 32;
    const int hi16 = lane >> 4, lo16 = lane & 15;

    // staging addresses for this thread
    const int arow = tid >> 1, aseg = (tid & 1) * 16;   // A: 128x32, 32B/thread
    const int brow = tid >> 2, bseg = (tid & 3) * 8;    // B: 64x32, 16B/thread
    const unsigned short* Ap  = A  + (size_t)arow * D + aseg;
    const unsigned short* B2p = B2 + (size_t)brow * D + bseg;
    const unsigned short* B3p = B3 + (size_t)brow * D + bseg;

    auto stage = [&](int k0, int s) {
#if MOE_ASYNC
        __builtin_amdgcn_global_load_async_to_lds_b128(
            AS1CAST(Ap + k0),  AS3CAST(&sA[s][arow * LDSPITCH + aseg]), 0, 0);
        __builtin_amdgcn_global_load_async_to_lds_b128(
            AS1CAST(Ap + k0),  AS3CAST(&sA[s][arow * LDSPITCH + aseg]), 16, 0);
        __builtin_amdgcn_global_load_async_to_lds_b128(
            AS1CAST(B2p + k0), AS3CAST(&sB2[s][brow * LDSPITCH + bseg]), 0, 0);
        __builtin_amdgcn_global_load_async_to_lds_b128(
            AS1CAST(B3p + k0), AS3CAST(&sB3[s][brow * LDSPITCH + bseg]), 0, 0);
#else
        const uint4* sa = (const uint4*)(Ap + k0);
        uint4 p0 = sa[0], p1 = sa[1];
        uint4* da = (uint4*)&sA[s][arow * LDSPITCH + aseg];
        da[0] = p0; da[1] = p1;
        *(uint4*)&sB2[s][brow * LDSPITCH + bseg] = *(const uint4*)(B2p + k0);
        *(uint4*)&sB3[s][brow * LDSPITCH + bseg] = *(const uint4*)(B3p + k0);
#endif
    };

    v8f acc2[2][2], acc3[2][2];
    const v8f vz = {0.f, 0.f, 0.f, 0.f, 0.f, 0.f, 0.f, 0.f};
#pragma unroll
    for (int mi = 0; mi < 2; ++mi)
#pragma unroll
        for (int ni = 0; ni < 2; ++ni) { acc2[mi][ni] = vz; acc3[mi][ni] = vz; }

    stage(0, 0);
    int cur = 0;
    for (int k0 = 0; k0 < D; k0 += 32) {
        moe_wait_async0();           // my stage-`cur` copies landed (no-op in sync path)
        __syncthreads();             // everyone's landed; buffer cur^1 fully drained
        if (k0 + 32 < D) stage(k0 + 32, cur ^ 1);

        v16bf a[2], b2[2], b3[2];
#pragma unroll
        for (int mi = 0; mi < 2; ++mi) a[mi] = frag_a(sA[cur], mw + mi * 16 + lo16, hi16);
#pragma unroll
        for (int ni = 0; ni < 2; ++ni) {
            b2[ni] = frag_b(sB2[cur], nw + ni * 16 + lo16, hi16);
            b3[ni] = frag_b(sB3[cur], nw + ni * 16 + lo16, hi16);
        }
#pragma unroll
        for (int mi = 0; mi < 2; ++mi)
#pragma unroll
            for (int ni = 0; ni < 2; ++ni) {
                acc2[mi][ni] = __builtin_amdgcn_wmma_f32_16x16x32_bf16(
                    false, a[mi], false, b2[ni], (short)0, acc2[mi][ni], false, false);
                acc3[mi][ni] = __builtin_amdgcn_wmma_f32_16x16x32_bf16(
                    false, a[mi], false, b3[ni], (short)0, acc3[mi][ni], false, false);
            }
        cur ^= 1;
    }
    // Epilogue: h = silu(acc2) * acc3, bf16 store. C layout: M = r + 8*hi16, N = lo16.
    unsigned short* out = hbuf + (size_t)e * cap * H;
#pragma unroll
    for (int mi = 0; mi < 2; ++mi)
#pragma unroll
        for (int ni = 0; ni < 2; ++ni)
#pragma unroll
            for (int r = 0; r < 8; ++r) {
                float v2 = acc2[mi][ni][r];
                float v3 = acc3[mi][ni][r];
                float hv = fast_silu(v2) * v3;
                int row = mBase + mw + mi * 16 + r + hi16 * 8;
                int col = nBase + nw + ni * 16 + lo16;
                out[(size_t)row * H + col] = f2bf_bits(hv);
            }
}

// ---------------------------------------------------------------------------
// 6) GEMM2: bout = h @ W1.  A: cap x H bf16, B (pre-transposed): D x H bf16.
// ---------------------------------------------------------------------------
__global__ __launch_bounds__(256) void smoe_ffn2(const unsigned short* __restrict__ hbuf,
                                                 const unsigned short* __restrict__ w1t,
                                                 float* __restrict__ bout,
                                                 int H, int D, int cap) {
    __shared__ __align__(16) unsigned short sA[2][128 * LDSPITCH];
    __shared__ __align__(16) unsigned short sB[2][64 * LDSPITCH];
    const int e = blockIdx.z;
    const int mBase = blockIdx.x * 128;
    const int nBase = blockIdx.y * 64;
    const unsigned short* A = hbuf + ((size_t)e * cap + mBase) * H;
    const unsigned short* B = w1t + ((size_t)e * D + nBase) * H;
    const int tid = threadIdx.x, lane = tid & 31, wv = tid >> 5;
    const int mw = (wv & 3) * 32, nw = (wv >> 2) * 32;
    const int hi16 = lane >> 4, lo16 = lane & 15;

    const int arow = tid >> 1, aseg = (tid & 1) * 16;
    const int brow = tid >> 2, bseg = (tid & 3) * 8;
    const unsigned short* Ap = A + (size_t)arow * H + aseg;
    const unsigned short* Bp = B + (size_t)brow * H + bseg;

    auto stage = [&](int k0, int s) {
#if MOE_ASYNC
        __builtin_amdgcn_global_load_async_to_lds_b128(
            AS1CAST(Ap + k0), AS3CAST(&sA[s][arow * LDSPITCH + aseg]), 0, 0);
        __builtin_amdgcn_global_load_async_to_lds_b128(
            AS1CAST(Ap + k0), AS3CAST(&sA[s][arow * LDSPITCH + aseg]), 16, 0);
        __builtin_amdgcn_global_load_async_to_lds_b128(
            AS1CAST(Bp + k0), AS3CAST(&sB[s][brow * LDSPITCH + bseg]), 0, 0);
#else
        const uint4* sa = (const uint4*)(Ap + k0);
        uint4 p0 = sa[0], p1 = sa[1];
        uint4* da = (uint4*)&sA[s][arow * LDSPITCH + aseg];
        da[0] = p0; da[1] = p1;
        *(uint4*)&sB[s][brow * LDSPITCH + bseg] = *(const uint4*)(Bp + k0);
#endif
    };

    v8f acc[2][2];
    const v8f vz = {0.f, 0.f, 0.f, 0.f, 0.f, 0.f, 0.f, 0.f};
#pragma unroll
    for (int mi = 0; mi < 2; ++mi)
#pragma unroll
        for (int ni = 0; ni < 2; ++ni) acc[mi][ni] = vz;

    stage(0, 0);
    int cur = 0;
    for (int k0 = 0; k0 < H; k0 += 32) {
        moe_wait_async0();
        __syncthreads();
        if (k0 + 32 < H) stage(k0 + 32, cur ^ 1);

        v16bf a[2], b[2];
#pragma unroll
        for (int mi = 0; mi < 2; ++mi) a[mi] = frag_a(sA[cur], mw + mi * 16 + lo16, hi16);
#pragma unroll
        for (int ni = 0; ni < 2; ++ni) b[ni] = frag_b(sB[cur], nw + ni * 16 + lo16, hi16);
#pragma unroll
        for (int mi = 0; mi < 2; ++mi)
#pragma unroll
            for (int ni = 0; ni < 2; ++ni)
                acc[mi][ni] = __builtin_amdgcn_wmma_f32_16x16x32_bf16(
                    false, a[mi], false, b[ni], (short)0, acc[mi][ni], false, false);
        cur ^= 1;
    }
    float* out = bout + (size_t)e * cap * D;
#pragma unroll
    for (int mi = 0; mi < 2; ++mi)
#pragma unroll
        for (int ni = 0; ni < 2; ++ni)
#pragma unroll
            for (int r = 0; r < 8; ++r) {
                int row = mBase + mw + mi * 16 + r + hi16 * 8;
                int col = nBase + nw + ni * 16 + lo16;
                out[(size_t)row * D + col] = acc[mi][ni][r];
            }
}

// ---------------------------------------------------------------------------
// 7) Gather + gate. One wave per token; zeros for dropped tokens.
// ---------------------------------------------------------------------------
__global__ __launch_bounds__(256) void smoe_gather(const float* __restrict__ bout,
                                                   const int* __restrict__ eid,
                                                   const int* __restrict__ slot,
                                                   const float* __restrict__ gate,
                                                   float* __restrict__ out,
                                                   int N, int D, int cap) {
    const int lane  = threadIdx.x & 31;
    const int token = blockIdx.x * 8 + (threadIdx.x >> 5);
    if (token >= N) return;
    const int s = slot[token];
    float* dst = out + (size_t)token * D;
    if (s < 0) {
        const float4 z = {0.f, 0.f, 0.f, 0.f};
        for (int j = lane * 4; j < D; j += 128) *(float4*)(dst + j) = z;
        return;
    }
    const int e = eid[token];
    const float g = gate[token];
    const float* src = bout + ((size_t)e * cap + s) * D;
    for (int j = lane * 4; j < D; j += 128) {
        float4 v = *(const float4*)(src + j);
        v.x *= g; v.y *= g; v.z *= g; v.w *= g;
        *(float4*)(dst + j) = v;
    }
}

// ---------------------------------------------------------------------------
// Host-side launch
// ---------------------------------------------------------------------------
static inline size_t align256(size_t x) { return (x + 255) & ~(size_t)255; }

extern "C" void kernel_launch(void* const* d_in, const int* in_sizes, int n_in,
                              void* d_out, int out_size, void* d_ws, size_t ws_size,
                              hipStream_t stream) {
    const float* x  = (const float*)d_in[0];
    const float* Wg = (const float*)d_in[1];
    const float* bg = (const float*)d_in[2];
    const float* W1 = (const float*)d_in[3];
    const float* W2 = (const float*)d_in[4];
    const float* W3 = (const float*)d_in[5];

    const int E = in_sizes[2];                 // 8
    const int D = in_sizes[1] / E;             // 1024
    const int N = in_sizes[0] / D;             // 8192
    const int H = in_sizes[3] / (E * D);       // 2048
    const int cap = (N / E) * 5 / 4;           // 1280  (CAPACITY_FACTOR=1.25)

    // Workspace layout
    size_t off = 0;
    int*            ws_eid  = (int*)((char*)d_ws + off);            off = align256(off + (size_t)N * 4);
    int*            ws_slot = (int*)((char*)d_ws + off);            off = align256(off + (size_t)N * 4);
    float*          ws_gate = (float*)((char*)d_ws + off);          off = align256(off + (size_t)N * 4);
    unsigned short* ws_buf  = (unsigned short*)((char*)d_ws + off); off = align256(off + (size_t)E * cap * D * 2);
    unsigned short* ws_h    = (unsigned short*)((char*)d_ws + off); off = align256(off + (size_t)E * cap * H * 2);
    float*          ws_bout = (float*)((char*)d_ws + off);          off = align256(off + (size_t)E * cap * D * 4);
    unsigned short* ws_w2t  = (unsigned short*)((char*)d_ws + off); off = align256(off + (size_t)E * D * H * 2);
    unsigned short* ws_w3t  = (unsigned short*)((char*)d_ws + off); off = align256(off + (size_t)E * D * H * 2);
    unsigned short* ws_w1t  = (unsigned short*)((char*)d_ws + off); off = align256(off + (size_t)E * D * H * 2);
    (void)ws_size; (void)n_in; (void)out_size;

    // 1) weight transpose+convert (bf16).  W2,W3: (D,H)->(H,D); W1: (H,D)->(D,H)
    smoe_cvt_t<<<dim3(H / 32, D / 32, E), 256, 0, stream>>>(W2, ws_w2t, D, H);
    smoe_cvt_t<<<dim3(H / 32, D / 32, E), 256, 0, stream>>>(W3, ws_w3t, D, H);
    smoe_cvt_t<<<dim3(D / 32, H / 32, E), 256, 0, stream>>>(W1, ws_w1t, H, D);

    // 2) routing
    smoe_route<<<dim3((N + 7) / 8), 256, 0, stream>>>(x, Wg, bg, ws_eid, ws_gate, N, D);

    // 3) deterministic capacity scan (N must be a multiple of 1024: 8192 here)
    smoe_scan<<<dim3(1), 1024, 0, stream>>>(ws_eid, ws_slot, N, cap);

    // 4) scatter tokens to expert buffers (bf16)
    smoe_scatter<<<dim3((N + 7) / 8), 256, 0, stream>>>(x, ws_eid, ws_slot, ws_buf, N, D, cap);

    // 5) expert GEMM1 (SwiGLU) and 6) expert GEMM2
    smoe_ffn1<<<dim3(cap / 128, H / 64, E), 256, 0, stream>>>(ws_buf, ws_w2t, ws_w3t, ws_h, D, H, cap);
    smoe_ffn2<<<dim3(cap / 128, D / 64, E), 256, 0, stream>>>(ws_h, ws_w1t, ws_bout, H, D, cap);

    // 7) gather + gate
    smoe_gather<<<dim3((N + 7) / 8), 256, 0, stream>>>(ws_bout, ws_eid, ws_slot, ws_gate,
                                                       (float*)d_out, N, D, cap);
}
